// LstmAgent_2319282340170
// MI455X (gfx1250) — matrix operand order repaired and checked by reading, not compiled, not run on hardware
//
#include <hip/hip_runtime.h>

#define TT   1024
#define BB   256
#define NN   (TT * BB)      // 262144 rows
#define OBS  128
#define HID  128
#define LIN  512
#define NACT 16

typedef _Float16 f16;
typedef __attribute__((ext_vector_type(16))) _Float16 v16h;
typedef __attribute__((ext_vector_type(8)))  _Float16 v8h;
typedef __attribute__((ext_vector_type(8)))  float    v8f;

__device__ __forceinline__ float sigm_fast(float x) {
    return __builtin_amdgcn_rcpf(1.0f + __expf(-x));
}
__device__ __forceinline__ float tanh_fast(float x) {
    return 2.0f * sigm_fast(2.0f * x) - 1.0f;
}

// ---------------------------------------------------------------------------
// Small prep kernels
// ---------------------------------------------------------------------------
__global__ void cvt_f32_to_f16(const float* __restrict__ s, f16* __restrict__ d, int n) {
    int i = blockIdx.x * blockDim.x + threadIdx.x;
    if (i < n) d[i] = (f16)s[i];
}

// dst[r*dld + c] = (f16)src[r*scols + c]   (scatter x into cat[:,128:256])
__global__ void cvt_f32_to_f16_strided(const float* __restrict__ s, f16* __restrict__ d,
                                       int scols, int dld, int n) {
    int i = blockIdx.x * blockDim.x + threadIdx.x;
    if (i < n) {
        int r = i / scols, c = i - r * scols;
        d[(size_t)r * dld + c] = (f16)s[i];
    }
}

__global__ void add_bias_vec(const float* __restrict__ a, const float* __restrict__ b,
                             float* __restrict__ o, int n) {
    int i = blockIdx.x * blockDim.x + threadIdx.x;
    if (i < n) o[i] = a[i] + b[i];
}

__global__ void pad_cri_w3(const float* __restrict__ w, f16* __restrict__ d) {
    int i = blockIdx.x * blockDim.x + threadIdx.x;
    if (i < 16 * 64) d[i] = (f16)((i < 64) ? w[i] : 0.0f);
}
__global__ void pad_cri_b3(const float* __restrict__ b, float* __restrict__ d) {
    int i = threadIdx.x;
    if (i < 16) d[i] = (i == 0) ? b[0] : 0.0f;
}

// ---------------------------------------------------------------------------
// WMMA GEMM:  C[M=NN, 16*NT per wave] = act( A(f16)[NN,K](lda) @ W(f16)[Nout,K]^T + bias )
// Each wave: 1 m-tile x NT n-tiles (A-fragment reused NT times, NT independent
// accumulator chains). 8 waves/block => block covers 128 rows.
// FRAG=1: f32 output stored in raw WMMA fragment layout
//         [m-tile][n-tile 0..ntn-1][lane][8] (8 contiguous floats per lane).
// grid = (Nout/(16*NT), NN/128)
// ---------------------------------------------------------------------------
template <int NT, int FRAG>
__global__ __launch_bounds__(256) void gemm_f16_wmma(
    const f16* __restrict__ A, int lda,
    const f16* __restrict__ W,          // [Nout, K] row-major
    const float* __restrict__ bias,     // [Nout] or null
    f16* __restrict__ C16, int ldc16,   // optional f16 output (row-major)
    float* __restrict__ C32, int ldc32, // optional f32 output (row-major or frag)
    int K, int act, int ntn)            // ntn = Nout/16 (frag layout stride)
{
    const int lane = threadIdx.x & 31;
    const int wv   = threadIdx.x >> 5;
    const int m0   = (blockIdx.y * 8 + wv) * 16;
    const int n0   = blockIdx.x * (16 * NT);
    const int l15  = lane & 15;
    const int hl   = lane >> 4;

    const f16* Ap = A + (size_t)(m0 + l15) * lda + hl * 8;   // A-frag row, K-half base
    const f16* Wp[NT];
    #pragma unroll
    for (int j = 0; j < NT; ++j)
        Wp[j] = W + (size_t)(n0 + 16 * j + l15) * K + hl * 8;

    v8f acc[NT];
    #pragma unroll
    for (int j = 0; j < NT; ++j)
        #pragma unroll
        for (int r = 0; r < 8; ++r) acc[j][r] = 0.0f;

    for (int k0 = 0; k0 < K; k0 += 32) {
        v16h a;
        *((v8h*)&a)   = *(const v8h*)(Ap + k0);        // K = base+0..7   (or 8..15)
        ((v8h*)&a)[1] = *(const v8h*)(Ap + k0 + 16);   // K = base+16..23 (or 24..31)
        #pragma unroll
        for (int j = 0; j < NT; ++j) {
            v16h b;
            *((v8h*)&b)   = *(const v8h*)(Wp[j] + k0);
            ((v8h*)&b)[1] = *(const v8h*)(Wp[j] + k0 + 16);
            acc[j] = __builtin_amdgcn_wmma_f32_16x16x32_f16(false, a, false, b,
                                                            (short)0, acc[j], false, false);
        }
    }

    const int mb = m0 + 8 * hl;
    #pragma unroll
    for (int j = 0; j < NT; ++j) {
        const int col = n0 + 16 * j + l15;
        const float bv = bias ? bias[col] : 0.0f;
        if (FRAG) {
            // 8 contiguous floats per lane -> 2x global_store_b128
            float* p = C32 + (((size_t)(m0 >> 4) * ntn + (n0 >> 4) + j) * 32 + lane) * 8;
            #pragma unroll
            for (int r = 0; r < 8; ++r) p[r] = acc[j][r] + bv;
        } else {
            #pragma unroll
            for (int r = 0; r < 8; ++r) {
                float v = acc[j][r] + bv;
                if (act) v = tanh_fast(v);
                if (C16) C16[(size_t)(mb + r) * ldc16 + col] = (f16)v;
                if (C32) C32[(size_t)(mb + r) * ldc32 + col] = v;
            }
        }
    }
}

// ---------------------------------------------------------------------------
// LSTM scan. 16 blocks x 16 batch rows; 8 waves, wave w owns hidden cols
// [16w,16w+16) across all 4 gates. w_hh fragments pinned in VGPRs for all
// 1024 steps; h kept in LDS as f16 (exact 0/1 done-mask applied at read);
// c kept f32 in registers in the native C-matrix layout. Gate pre-sums are
// read in fragment layout (2x b128 per gate per lane) and the next step's
// contiguous 32KB slice is prefetched (global_prefetch_b8).
// ---------------------------------------------------------------------------
__global__ __launch_bounds__(256) void lstm_scan(
    const float* __restrict__ gatesF,  // frag layout [NN/16][32][32 lanes][8]
    const float* __restrict__ done_,   // [T*B]
    const float* __restrict__ h0,      // [B,128]
    const float* __restrict__ c0,      // [B,128]
    const f16*  __restrict__ whh,      // [512,128] f16
    f16*  __restrict__ cat,            // [T*B,256]; cols 0..127 receive h
    float* __restrict__ hT,            // [B,128]
    float* __restrict__ cT)            // [B,128]
{
    __shared__ f16 h_s[16][136];       // 272B row stride: 16 rows hit 16 distinct banks

    const int lane = threadIdx.x & 31;
    const int wv   = threadIdx.x >> 5;
    const int l15  = lane & 15;
    const int hl   = lane >> 4;
    const int b0   = blockIdx.x * 16;
    const int col  = wv * 16 + l15;    // owned hidden column (C-layout)
    const int mrow = 8 * hl;           // C-layout row base

    // Pin w_hh B-fragments in registers: 4 gates x 4 k-steps
    v16h bf[4][4];
    #pragma unroll
    for (int g = 0; g < 4; ++g) {
        const f16* wp = whh + (size_t)(g * 128 + wv * 16 + l15) * 128 + hl * 8;
        #pragma unroll
        for (int kk = 0; kk < 4; ++kk) {
            *((v8h*)&bf[g][kk])   = *(const v8h*)(wp + kk * 32);
            ((v8h*)&bf[g][kk])[1] = *(const v8h*)(wp + kk * 32 + 16);
        }
    }

    // init h in LDS (f16), c in registers (f32)
    for (int i = threadIdx.x; i < 16 * 128; i += 256) {
        int r = i >> 7, k = i & 127;
        h_s[r][k] = (f16)h0[(size_t)(b0 + r) * 128 + k];
    }
    float creg[8];
    #pragma unroll
    for (int r = 0; r < 8; ++r)
        creg[r] = c0[(size_t)(b0 + mrow + r) * 128 + col];

    const int ka = hl * 8;
    for (int t = 0; t < TT; ++t) {
        const size_t mt = (size_t)t * 16 + blockIdx.x;   // this block's m-tile
        __syncthreads();   // h_s ready
        // prefetch next step's contiguous 32KB frag slice: one 128B line/thread
        if (t + 1 < TT)
            __builtin_prefetch(gatesF + (mt + 16) * 8192 + threadIdx.x * 32, 0, 3);

        // A-fragment of done-masked h (fragment row = l15; mask is exactly 0/1)
        const f16 mah = (f16)(1.0f - done_[(size_t)t * BB + b0 + l15]);
        v16h af[4];
        #pragma unroll
        for (int kk = 0; kk < 4; ++kk) {
            v8h lo = *(const v8h*)&h_s[l15][kk * 32 + ka];
            v8h hi = *(const v8h*)&h_s[l15][kk * 32 + ka + 16];
            lo = lo * mah;
            hi = hi * mah;
            *((v8h*)&af[kk])   = lo;
            ((v8h*)&af[kk])[1] = hi;
        }
        __syncthreads();   // all reads done before this step's h writes

        // gate pre-sums for this wave's 4 tiles: 2x b128 per gate per lane
        float gxl[4][8];
        #pragma unroll
        for (int g = 0; g < 4; ++g) {
            const float* p = gatesF + ((mt * 32 + (g * 8 + wv)) * 32 + lane) * 8;
            *(float4*)&gxl[g][0] = *(const float4*)p;
            *(float4*)&gxl[g][4] = *(const float4*)(p + 4);
        }

        v8f acc[4];
        #pragma unroll
        for (int g = 0; g < 4; ++g) {
            #pragma unroll
            for (int r = 0; r < 8; ++r) acc[g][r] = 0.0f;
            #pragma unroll
            for (int kk = 0; kk < 4; ++kk)
                acc[g] = __builtin_amdgcn_wmma_f32_16x16x32_f16(
                    false, af[kk], false, bf[g][kk], (short)0, acc[g], false, false);
        }

        #pragma unroll
        for (int r = 0; r < 8; ++r) {
            const int m = mrow + r;
            const float mc = 1.0f - done_[(size_t)t * BB + b0 + m];
            float ii = acc[0][r] + gxl[0][r];
            float ff = acc[1][r] + gxl[1][r];
            float gg = acc[2][r] + gxl[2][r];
            float oo = acc[3][r] + gxl[3][r];
            float cn = sigm_fast(ff) * (creg[r] * mc) + sigm_fast(ii) * tanh_fast(gg);
            float hn = sigm_fast(oo) * tanh_fast(cn);
            creg[r] = cn;
            h_s[m][col] = (f16)hn;                                   // next-step state
            cat[((size_t)t * BB + b0 + m) * 256 + col] = (f16)hn;    // hs output
            if (t == TT - 1) hT[(size_t)(b0 + m) * 128 + col] = hn;  // f32 carry out
        }
    }

    #pragma unroll
    for (int r = 0; r < 8; ++r)
        cT[(size_t)(b0 + mrow + r) * 128 + col] = creg[r];
}

// ---------------------------------------------------------------------------
// log-softmax / entropy / value gather
// ---------------------------------------------------------------------------
__global__ void finalize_out(const float* __restrict__ logits,  // [N,16]
                             const float* __restrict__ valpad,  // [N,16] (col 0 valid)
                             const int*   __restrict__ action,
                             float* __restrict__ out)
{
    int r = blockIdx.x * blockDim.x + threadIdx.x;
    if (r >= NN) return;
    const float* lg = logits + (size_t)r * 16;
    float mx = lg[0];
    #pragma unroll
    for (int i = 1; i < 16; ++i) mx = fmaxf(mx, lg[i]);
    float s = 0.0f;
    #pragma unroll
    for (int i = 0; i < 16; ++i) s += __expf(lg[i] - mx);
    float lse = mx + __logf(s);
    float ent = 0.0f;
    #pragma unroll
    for (int i = 0; i < 16; ++i) { float l = lg[i] - lse; ent -= __expf(l) * l; }
    out[(size_t)r * 3 + 0] = lg[action[r]] - lse;
    out[(size_t)r * 3 + 1] = ent;
    out[(size_t)r * 3 + 2] = valpad[(size_t)r * 16];
}

// ---------------------------------------------------------------------------
// Host
// ---------------------------------------------------------------------------
static void launch_gemm(const f16* A, int lda, const f16* W, const float* bias,
                        f16* C16, int ldc16, float* C32, int ldc32,
                        int K, int Nout, int act, int frag, hipStream_t s) {
    if (Nout % 64 == 0) {
        dim3 grid(Nout / 64, NN / 128);
        if (frag)
            gemm_f16_wmma<4, 1><<<grid, 256, 0, s>>>(A, lda, W, bias, C16, ldc16, C32, ldc32,
                                                     K, act, Nout / 16);
        else
            gemm_f16_wmma<4, 0><<<grid, 256, 0, s>>>(A, lda, W, bias, C16, ldc16, C32, ldc32,
                                                     K, act, Nout / 16);
    } else {
        dim3 grid(Nout / 16, NN / 128);
        gemm_f16_wmma<1, 0><<<grid, 256, 0, s>>>(A, lda, W, bias, C16, ldc16, C32, ldc32,
                                                 K, act, Nout / 16);
    }
}

extern "C" void kernel_launch(void* const* d_in, const int* in_sizes, int n_in,
                              void* d_out, int out_size, void* d_ws, size_t ws_size,
                              hipStream_t stream) {
    (void)in_sizes; (void)n_in; (void)out_size; (void)ws_size;
    const float* x      = (const float*)d_in[0];
    const float* done_  = (const float*)d_in[1];
    const float* h0     = (const float*)d_in[2];
    const float* c0     = (const float*)d_in[3];
    const int*   action = (const int*)  d_in[4];
    const float* net_w1 = (const float*)d_in[5];  const float* net_b1 = (const float*)d_in[6];
    const float* net_w2 = (const float*)d_in[7];  const float* net_b2 = (const float*)d_in[8];
    const float* net_w3 = (const float*)d_in[9];  const float* net_b3 = (const float*)d_in[10];
    const float* w_ih   = (const float*)d_in[11]; const float* b_ih   = (const float*)d_in[12];
    const float* w_hh   = (const float*)d_in[13]; const float* b_hh   = (const float*)d_in[14];
    const float* act_w1 = (const float*)d_in[15]; const float* act_b1 = (const float*)d_in[16];
    const float* act_w2 = (const float*)d_in[17]; const float* act_b2 = (const float*)d_in[18];
    const float* act_w3 = (const float*)d_in[19]; const float* act_b3 = (const float*)d_in[20];
    const float* cri_w1 = (const float*)d_in[21]; const float* cri_b1 = (const float*)d_in[22];
    const float* cri_w2 = (const float*)d_in[23]; const float* cri_b2 = (const float*)d_in[24];
    const float* cri_w3 = (const float*)d_in[25]; const float* cri_b3 = (const float*)d_in[26];

    char* ws = (char*)d_ws;
    size_t o = 0;
    auto alloc = [&](size_t bytes) { size_t r = o; o += (bytes + 255) & ~(size_t)255; return r; };
    const size_t o_cat  = alloc((size_t)NN * 256 * 2);   // cat f16 [N,256]
    const size_t o_h1   = alloc((size_t)NN * 64 * 2);
    const size_t o_h2   = alloc((size_t)NN * 64 * 2);
    const size_t o_feat = alloc((size_t)NN * 512 * 2);   // reused for a1/a2/c1/c2 after gates GEMM
    const size_t o_gate = alloc((size_t)NN * 512 * 4);   // frag layout; reused for logits/valpad
    const size_t o_nw1 = alloc(64*128*2), o_nw2 = alloc(64*64*2), o_nw3 = alloc(512*64*2);
    const size_t o_wih = alloc(512*512*2), o_whh = alloc(512*128*2);
    const size_t o_aw1 = alloc(64*256*2), o_aw2 = alloc(64*64*2), o_aw3 = alloc(16*64*2);
    const size_t o_cw1 = alloc(64*256*2), o_cw2 = alloc(64*64*2), o_cw3 = alloc(16*64*2);
    const size_t o_bsum = alloc(512*4), o_b3p = alloc(16*4);

    f16*  cat   = (f16*)(ws + o_cat);
    f16*  h1b   = (f16*)(ws + o_h1);
    f16*  h2b   = (f16*)(ws + o_h2);
    f16*  feat  = (f16*)(ws + o_feat);
    float* gates = (float*)(ws + o_gate);
    f16*  a1  = feat;                        // aliases (feat dead after gates GEMM)
    f16*  a2  = feat + (size_t)NN * 64;
    f16*  cc1 = feat + (size_t)NN * 64 * 2;
    f16*  cc2 = feat + (size_t)NN * 64 * 3;
    float* logits = gates;                   // aliases (gates dead after LSTM)
    float* valpad = (float*)((char*)gates + (size_t)NN * 16 * 4);
    f16* nw1=(f16*)(ws+o_nw1); f16* nw2=(f16*)(ws+o_nw2); f16* nw3=(f16*)(ws+o_nw3);
    f16* wih=(f16*)(ws+o_wih); f16* whh=(f16*)(ws+o_whh);
    f16* aw1=(f16*)(ws+o_aw1); f16* aw2=(f16*)(ws+o_aw2); f16* aw3=(f16*)(ws+o_aw3);
    f16* cw1=(f16*)(ws+o_cw1); f16* cw2=(f16*)(ws+o_cw2); f16* cw3=(f16*)(ws+o_cw3);
    float* bsum=(float*)(ws+o_bsum); float* b3p=(float*)(ws+o_b3p);

    auto cvt = [&](const float* s, f16* d, int n) {
        cvt_f32_to_f16<<<(n + 255) / 256, 256, 0, stream>>>(s, d, n);
    };
    // weight conversions (tiny; stay in L2)
    cvt(net_w1, nw1, 64*128);  cvt(net_w2, nw2, 64*64);  cvt(net_w3, nw3, 512*64);
    cvt(w_ih,  wih, 512*512);  cvt(w_hh,  whh, 512*128);
    cvt(act_w1, aw1, 64*256);  cvt(act_w2, aw2, 64*64);  cvt(act_w3, aw3, 16*64);
    cvt(cri_w1, cw1, 64*256);  cvt(cri_w2, cw2, 64*64);
    pad_cri_w3<<<(16*64 + 255) / 256, 256, 0, stream>>>(cri_w3, cw3);
    pad_cri_b3<<<1, 16, 0, stream>>>(cri_b3, b3p);
    add_bias_vec<<<2, 256, 0, stream>>>(b_ih, b_hh, bsum, 512);
    // x -> cat[:,128:256] (f16); feature net reads it strided
    {
        int n = NN * OBS;
        cvt_f32_to_f16_strided<<<(n + 255) / 256, 256, 0, stream>>>(x, cat + 128, OBS, 256, n);
    }

    // feature net
    launch_gemm(cat + 128, 256, nw1, net_b1, h1b, 64, nullptr, 0, 128, 64, 1, 0, stream);
    launch_gemm(h1b, 64,  nw2, net_b2, h2b, 64, nullptr, 0, 64, 64, 1, 0, stream);
    launch_gemm(h2b, 64,  nw3, net_b3, feat, 512, nullptr, 0, 64, 512, 0, 0, stream);
    // gate pre-sums: feat @ w_ih^T + (b_ih + b_hh), f32 in WMMA fragment layout
    launch_gemm(feat, 512, wih, bsum, nullptr, 0, gates, 512, 512, 512, 0, 1, stream);

    // sequential LSTM scan: writes cat[:,0:128] and hT/cT tail of d_out
    float* outp = (float*)d_out;
    float* hT = outp + (size_t)NN * 3;
    float* cT = hT + (size_t)BB * HID;
    lstm_scan<<<16, 256, 0, stream>>>(gates, done_, h0, c0, whh, cat, hT, cT);

    // actor head
    launch_gemm(cat, 256, aw1, act_b1, a1, 64, nullptr, 0, 256, 64, 1, 0, stream);
    launch_gemm(a1, 64,  aw2, act_b2, a2, 64, nullptr, 0, 64, 64, 1, 0, stream);
    launch_gemm(a2, 64,  aw3, act_b3, nullptr, 0, logits, 16, 64, 16, 0, 0, stream);
    // critic head (padded to Nout=16)
    launch_gemm(cat, 256, cw1, cri_b1, cc1, 64, nullptr, 0, 256, 64, 1, 0, stream);
    launch_gemm(cc1, 64, cw2, cri_b2, cc2, 64, nullptr, 0, 64, 64, 1, 0, stream);
    launch_gemm(cc2, 64, cw3, b3p, nullptr, 0, valpad, 16, 64, 16, 0, 0, stream);

    finalize_out<<<(NN + 255) / 256, 256, 0, stream>>>(logits, valpad, action, outp);
}
